// PLI_GCN_Extractor_9423158248185
// MI455X (gfx1250) — matrix operand order repaired and checked by reading, not compiled, not run on hardware
//
#include <hip/hip_runtime.h>
#include <hip/hip_bf16.h>
#include <math.h>

typedef __attribute__((ext_vector_type(16))) _Float16     v16h;
typedef __attribute__((ext_vector_type(8)))  float        v8f;
typedef __attribute__((ext_vector_type(4)))  unsigned int v4u;
typedef __attribute__((ext_vector_type(8)))  int          v8i;
typedef __attribute__((ext_vector_type(4)))  int          v4i;

#define BATCH 128
#define CCH   64
#define TT    128
#define HID   128
#define EE    64

#if defined(__HIP_DEVICE_COMPILE__) && __has_builtin(__builtin_amdgcn_tensor_load_to_lds)
#define HAVE_TDM 1
#else
#define HAVE_TDM 0
#endif

// ---------------------------------------------------------------------------
// TDM: DMA a contiguous block (nbytes, multiple of 4) from global to LDS as a
// 1-D tile. D# built per CDNA5 ISA §8.3/§8.4:
//   group0: [1:0]=count=1, [63:32]=lds_addr, [120:64]=global_addr, [127:126]=2
//   group1: data_size=4B, tensor_dim0=tile_dim0=ndwords, stride0=ndwords
// Issue from ONE wave (TDM ignores EXEC; one issue per transfer), then drain
// TENSORcnt in the issuing wave before the workgroup barrier.
// ---------------------------------------------------------------------------
__device__ __forceinline__ void tdm_copy_to_lds(void* lds_dst, const void* gsrc,
                                                unsigned nbytes) {
#if HAVE_TDM
    const unsigned     ne  = nbytes >> 2;  // dwords
    unsigned           lof = (unsigned)__builtin_amdgcn_readfirstlane(
        (int)(unsigned)(uintptr_t)lds_dst);             // flat addr[31:0] == LDS offset
    unsigned long long ga  = (unsigned long long)(uintptr_t)gsrc;
    unsigned           glo = (unsigned)__builtin_amdgcn_readfirstlane(
        (int)(unsigned)(ga & 0xffffffffull));
    unsigned           ghi = (unsigned)__builtin_amdgcn_readfirstlane(
        (int)(unsigned)(ga >> 32));
    v4u g0;
    g0[0] = 1u;                                  // count=1, user descriptor
    g0[1] = lof;                                 // lds_addr
    g0[2] = glo;                                 // global_addr[31:0]
    g0[3] = (ghi & 0x01ffffffu) | (2u << 30);    // global_addr[56:32] | type=2
    v8i g1;
    g1[0] = (int)(2u << 16);                     // data_size = 4 bytes
    g1[1] = (int)((ne & 0xffffu) << 16);         // tensor_dim0[15:0]  @bit48
    g1[2] = (int)(((ne >> 16) & 0xffffu) | (1u << 16));  // dim0[31:16] | tensor_dim1=1
    g1[3] = (int)((ne & 0xffffu) << 16);         // tile_dim0 @bit112 (ne <= 65535)
    g1[4] = 1;                                   // tile_dim1 = 1
    g1[5] = (int)ne;                             // tensor_dim0_stride[31:0]
    g1[6] = 0;
    g1[7] = 0;
    v4i z4 = {};
#if __has_include(<hip/amd_detail/amd_gfx1250_TDM.h>)
    v8i z8 = {};
    __builtin_amdgcn_tensor_load_to_lds(g0, g1, z4, z4, z8, 0);   // clang-23 arity
#else
    __builtin_amdgcn_tensor_load_to_lds(g0, g1, z4, z4, 0);       // ROCm 7.2 arity
#endif
    __builtin_amdgcn_s_wait_tensorcnt(0);
#endif
}

// ---------------------------------------------------------------------------
// Generic cooperative WMMA GEMM on LDS operands.
//   A  : M x K, row-major (f16, LDS), leading dim lda
//   Bt : N x K, row-major (f16, LDS)  == B transposed, leading dim ldb
//   Kpad multiple of 32; elements with k >= kmax are treated as zero.
//   ep(m, n, val) invoked once per output element (8 per lane per tile).
// Executed by ALL threads of the block (EXEC all-ones for v_wmma).
// ---------------------------------------------------------------------------
template <typename Ep>
__device__ __forceinline__ void gemm_wmma(const _Float16* __restrict__ A, int lda,
                                          const _Float16* __restrict__ Bt, int ldb,
                                          int M, int N, int Kpad, int kmax, Ep ep) {
    const int lane   = threadIdx.x & 31;
    const int wave   = threadIdx.x >> 5;
    const int nWaves = blockDim.x >> 5;
    const int half   = lane >> 4;
    const int r16    = lane & 15;
    const int tilesN = N >> 4;
    const int tiles  = (M >> 4) * tilesN;
    for (int t = wave; t < tiles; t += nWaves) {
        const int mt = t / tilesN, nt = t % tilesN;
        v8f acc = {};
        for (int kk = 0; kk < Kpad; kk += 32) {
            v16h a, b;
            const int am = (mt * 16 + r16) * lda;
#pragma unroll
            for (int e = 0; e < 16; ++e) {
                // ISA 7.12.2: 16-bit A 16x32 layout
                int k = kk + ((e < 8) ? 0 : 16) + half * 8 + (e & 7);
                a[e] = (k < kmax) ? A[am + k] : (_Float16)0.0f;
            }
            const int bn = (nt * 16 + r16) * ldb;
#pragma unroll
            for (int e = 0; e < 16; ++e) {
                // B 32x16: same K layout as A (ISA 7.12.2 16-bit layout)
                int k = kk + ((e < 8) ? 0 : 16) + half * 8 + (e & 7);
                b[e] = (k < kmax) ? Bt[bn + k] : (_Float16)0.0f;
            }
            acc = __builtin_amdgcn_wmma_f32_16x16x32_f16(
                false, a, false, b, (short)0, acc, false, false);
        }
#pragma unroll
        for (int r = 0; r < 8; ++r)
            ep(mt * 16 + half * 8 + r, nt * 16 + r16, acc[r]);
    }
}

// ---------------------------------------------------------------------------
// K0: zero the BN statistics accumulators (256 floats at ws[0..1024))
// ---------------------------------------------------------------------------
__global__ __launch_bounds__(256) void k0_zero(float* stats) {
    if (threadIdx.x < 256) stats[threadIdx.x] = 0.0f;
}

// ---------------------------------------------------------------------------
// K1: per batch — Hilbert analytic signal, WPLI, normalized adjacency,
//     support1 = adj_n @ eeg (WMMA), y1 = support1 @ W1^T + b1 (WMMA),
//     per-channel BN1 statistics.
// ---------------------------------------------------------------------------
__global__ __launch_bounds__(256) void k1_wpli_gcn1(
    const float* __restrict__ eeg, const float* __restrict__ w1,
    const float* __restrict__ b1, float* __restrict__ out_wpli,
    _Float16* __restrict__ ws_adj, _Float16* __restrict__ ws_y1,
    float* __restrict__ stats1) {
    __shared__ char  smem[65536];
    __shared__ float kre[TT], kim[TT];
    __shared__ float rsq[CCH];
    __shared__ float sSum[CCH], sSq[CCH];

    float*    E32 = (float*)smem;                  // [64][128] f32   [0,32K)
    _Float16* RE  = (_Float16*)(smem + 32768);     // [64][128] f16   [32K,48K)
    _Float16* IM  = (_Float16*)(smem + 49152);     // [64][128] f16   [48K,64K)
    float*    PLI = (float*)smem;                  // [64][64]  f32   [0,16K)  (over E32)
    _Float16* ADJ = (_Float16*)(smem + 16384);     // [64][64]  f16   [16K,24K)
    _Float16* ET  = (_Float16*)(smem + 24576);     // [128][64] f16   [24K,40K)
    _Float16* S1  = (_Float16*)(smem + 40960);     // [64][128] f16   [40K,56K)
    _Float16* W1  = (_Float16*)smem;               // [128][128] f16  [0,32K)  (late)

    const int    b   = blockIdx.x;
    const int    tid = threadIdx.x;
    const float* eb  = eeg + (size_t)b * CCH * TT;
    __builtin_prefetch(w1, 0, 1);

    // stage 0: eeg -> LDS (TDM DMA), Hilbert circular-conv kernel K[d]
#if HAVE_TDM
    if ((tid >> 5) == 0) tdm_copy_to_lds(E32, eb, CCH * TT * 4);
#else
    for (int i = tid; i < CCH * TT; i += 256) E32[i] = eb[i];
#endif
    if (tid < TT) {
        float sre = 0.f, sim = 0.f;
        for (int k = 0; k < 64; ++k) {
            float th = (float)M_PI * (float)k * (float)tid * (1.0f / 64.0f);
            sre += __cosf(th);
            sim += __sinf(th);
        }
        kre[tid] = sre * (1.0f / 64.0f);
        kim[tid] = sim * (1.0f / 64.0f);
    }
    if (tid < CCH) { sSum[tid] = 0.f; sSq[tid] = 0.f; }
    __syncthreads();

    // stage 1: analytic signal (Re, Im) per (channel, t)
    for (int i = tid; i < CCH * TT; i += 256) {
        int c = i >> 7, t = i & 127;
        const float* row = E32 + c * TT;
        float re = 0.f, im = 0.f;
        for (int s = 0; s < TT; ++s) {
            int   d = (t - s) & 127;
            float x = row[s];
            re += x * kre[d];
            im += x * kim[d];
        }
        RE[i] = (_Float16)re;
        IM[i] = (_Float16)im;
    }
    __syncthreads();

    // stage 2: WPLI = |mean_t sign(Im_i Re_j - Re_i Im_j)|, zero diagonal
    for (int p = tid; p < CCH * CCH; p += 256) {
        int   i = p >> 6, j = p & 63;
        float v = 0.f;
        if (i != j) {
            const _Float16* ri = RE + i * TT; const _Float16* ii = IM + i * TT;
            const _Float16* rj = RE + j * TT; const _Float16* ij = IM + j * TT;
            int s = 0;
            for (int t = 0; t < TT; ++t) {
                float cr = (float)ii[t] * (float)rj[t] - (float)ri[t] * (float)ij[t];
                s += (cr > 0.f) - (cr < 0.f);
            }
            v = fabsf((float)s) * (1.0f / (float)TT);
        }
        PLI[p] = v;
        out_wpli[(size_t)b * CCH * CCH + p] = v;
    }
    __syncthreads();

    // degree^{-1/2}
    if (tid < CCH) {
        float d = 0.f;
        for (int j = 0; j < CCH; ++j) d += PLI[tid * CCH + j];
        rsq[tid] = rsqrtf(d + 1e-8f);
    }
    __syncthreads();

    // normalized adjacency (f16, symmetric) + eeg^T (f16, re-read from L2)
    for (int p = tid; p < CCH * CCH; p += 256) {
        int      i = p >> 6, j = p & 63;
        _Float16 v = (_Float16)(PLI[p] * rsq[i] * rsq[j]);
        ADJ[p]                            = v;
        ws_adj[(size_t)b * CCH * CCH + p] = v;
    }
    for (int i = tid; i < CCH * TT; i += 256) {
        int c = i >> 7, t = i & 127;
        ET[t * CCH + c] = (_Float16)eb[c * TT + t];
    }
    __syncthreads();

    // support1 = adj_n @ eeg : A=ADJ[64x64], Bt=ET[128x64]
    gemm_wmma(ADJ, 64, ET, 64, 64, 128, 64, 64,
              [&](int m, int n, float v) { S1[m * 128 + n] = (_Float16)v; });
    __syncthreads();

    for (int i = tid; i < HID * TT; i += 256) W1[i] = (_Float16)w1[i];
    __syncthreads();

    // y1 = support1 @ W1^T + b1 ; W1 is already [out=128][in=128] = Bt
    _Float16* y1b = ws_y1 + (size_t)b * CCH * HID;
    gemm_wmma(S1, 128, W1, 128, 64, 128, 128, 128,
              [&](int m, int n, float v) {
                  v += b1[n];
                  y1b[m * HID + n] = (_Float16)v;
                  atomicAdd(&sSum[m], v);
                  atomicAdd(&sSq[m], v * v);
              });
    __syncthreads();
    if (tid < CCH) {
        atomicAdd(&stats1[tid], sSum[tid]);
        atomicAdd(&stats1[CCH + tid], sSq[tid]);
    }
}

// ---------------------------------------------------------------------------
// K2: BN1 + ReLU, support2 = adj_n @ x1 (WMMA), y2 = support2 @ W2^T + b2,
//     per-channel BN2 statistics.  Adjacency staged in via TDM.
// ---------------------------------------------------------------------------
__global__ __launch_bounds__(256) void k2_gcn2(
    const float* __restrict__ w2, const float* __restrict__ b2,
    const float* __restrict__ bn1g, const float* __restrict__ bn1b,
    const _Float16* __restrict__ ws_adj, const _Float16* __restrict__ ws_y1,
    _Float16* __restrict__ ws_y2, const float* __restrict__ stats1,
    float* __restrict__ stats2) {
    __shared__ _Float16 X1T[TT * CCH];   // [128][64] x1^T
    __shared__ _Float16 ADJ[CCH * CCH];
    __shared__ _Float16 S2[CCH * HID];
    __shared__ _Float16 W2[EE * HID];    // [64][128] = Bt
    __shared__ float    aSc[CCH], aSh[CCH], sSum[CCH], sSq[CCH];

    const int b = blockIdx.x, tid = threadIdx.x;
    __builtin_prefetch(w2, 0, 1);
#if HAVE_TDM
    if ((tid >> 5) == 0)
        tdm_copy_to_lds(ADJ, ws_adj + (size_t)b * CCH * CCH, CCH * CCH * 2);
#else
    for (int i = tid; i < CCH * CCH; i += 256) ADJ[i] = ws_adj[(size_t)b * CCH * CCH + i];
#endif
    if (tid < CCH) {
        float n    = (float)(BATCH * HID);
        float mean = stats1[tid] / n;
        float var  = stats1[CCH + tid] / n - mean * mean;
        float inv  = rsqrtf(var + 1e-5f);
        aSc[tid]   = inv * bn1g[tid];
        aSh[tid]   = bn1b[tid] - mean * inv * bn1g[tid];
        sSum[tid]  = 0.f;
        sSq[tid]   = 0.f;
    }
    __syncthreads();

    const _Float16* y1b = ws_y1 + (size_t)b * CCH * HID;
    for (int i = tid; i < CCH * HID; i += 256) {
        int   c = i >> 7, f = i & 127;
        float v = fmaxf((float)y1b[i] * aSc[c] + aSh[c], 0.f);
        X1T[f * CCH + c] = (_Float16)v;
    }
    for (int i = tid; i < EE * HID; i += 256) W2[i] = (_Float16)w2[i];
    __syncthreads();

    gemm_wmma(ADJ, 64, X1T, 64, 64, 128, 64, 64,
              [&](int m, int n, float v) { S2[m * 128 + n] = (_Float16)v; });
    __syncthreads();

    _Float16* y2b = ws_y2 + (size_t)b * CCH * EE;
    gemm_wmma(S2, 128, W2, 128, 64, 64, 128, 128,
              [&](int m, int n, float v) {
                  v += b2[n];
                  y2b[m * EE + n] = (_Float16)v;
                  atomicAdd(&sSum[m], v);
                  atomicAdd(&sSq[m], v * v);
              });
    __syncthreads();
    if (tid < CCH) {
        atomicAdd(&stats2[tid], sSum[tid]);
        atomicAdd(&stats2[CCH + tid], sSq[tid]);
    }
}

// ---------------------------------------------------------------------------
// K3: BN2 + ReLU, fused 4-head attention (all GEMMs via WMMA), out-proj,
//     token-mean -> features.
// ---------------------------------------------------------------------------
__global__ __launch_bounds__(256) void k3_mha(
    const float* __restrict__ winw, const float* __restrict__ winb,
    const float* __restrict__ woutw, const float* __restrict__ woutb,
    const float* __restrict__ bn2g, const float* __restrict__ bn2b,
    const _Float16* __restrict__ ws_y2, const float* __restrict__ stats2,
    float* __restrict__ feat) {
    __shared__ char  smem[65536];
    __shared__ float aSc[CCH], aSh[CCH], fmean[EE];

    _Float16* X2  = (_Float16*)smem;               // [64][64]   [0,8K)
    _Float16* WIN = (_Float16*)(smem + 8192);      // [192][64]  [8K,32K)
    _Float16* QKV = (_Float16*)(smem + 32768);     // [64][192]  [32K,56K)
    _Float16* O   = (_Float16*)(smem + 57344);     // [64][64]   [56K,64K)
    _Float16* VT  = (_Float16*)smem;               // [4][16][64][0,8K)   (over X2)
    float*    SC  = (float*)(smem + 8192);         // [64][64]f32[8K,24K) (over WIN)
    _Float16* AT  = (_Float16*)(smem + 24576);     // [64][64]   [24K,32K)
    _Float16* WO  = (_Float16*)(smem + 8192);      // [64][64]   [8K,16K) (over SC, late)

    const int b = blockIdx.x, tid = threadIdx.x;
    __builtin_prefetch(winw, 0, 1);
    __builtin_prefetch(woutw, 0, 1);
    if (tid < CCH) {
        float n    = (float)(BATCH * EE);
        float mean = stats2[tid] / n;
        float var  = stats2[CCH + tid] / n - mean * mean;
        float inv  = rsqrtf(var + 1e-5f);
        aSc[tid]   = inv * bn2g[tid];
        aSh[tid]   = bn2b[tid] - mean * inv * bn2g[tid];
        fmean[tid] = 0.f;
    }
    __syncthreads();

    const _Float16* y2b = ws_y2 + (size_t)b * CCH * EE;
    for (int i = tid; i < CCH * EE; i += 256) {
        int c = i >> 6;
        X2[i] = (_Float16)fmaxf((float)y2b[i] * aSc[c] + aSh[c], 0.f);
    }
    for (int i = tid; i < 3 * EE * EE; i += 256) WIN[i] = (_Float16)winw[i];
    __syncthreads();

    // qkv = x2 @ Win^T + b ; Win [192][64] already row-major Bt
    gemm_wmma(X2, 64, WIN, 64, 64, 192, 64, 64,
              [&](int m, int n, float v) { QKV[m * 192 + n] = (_Float16)(v + winb[n]); });
    __syncthreads();

    // v^T per head: VT[h][d][t] = QKV[t][128 + h*16 + d]
    for (int i = tid; i < 4 * 16 * 64; i += 256) {
        int h = i >> 10, d = (i >> 6) & 15, t = i & 63;
        VT[i] = QKV[t * 192 + 128 + h * 16 + d];
    }
    __syncthreads();

    for (int h = 0; h < 4; ++h) {
        // scores = q_h @ k_h^T * 0.25 ; strided views into QKV, K padded 16->32
        gemm_wmma(QKV + h * 16, 192, QKV + 64 + h * 16, 192, 64, 64, 32, 16,
                  [&](int m, int n, float v) { SC[m * 64 + n] = v * 0.25f; });
        __syncthreads();
        if (tid < 64) {  // row softmax
            float mx = -1e30f;
            for (int n = 0; n < 64; ++n) mx = fmaxf(mx, SC[tid * 64 + n]);
            float s = 0.f;
            for (int n = 0; n < 64; ++n) s += __expf(SC[tid * 64 + n] - mx);
            float r = 1.0f / s;
            for (int n = 0; n < 64; ++n)
                AT[tid * 64 + n] = (_Float16)(__expf(SC[tid * 64 + n] - mx) * r);
        }
        __syncthreads();
        // o_h = attn @ v_h : Bt = VT[h] (16x64)
        gemm_wmma(AT, 64, VT + h * 1024, 64, 64, 16, 64, 64,
                  [&](int m, int n, float v) { O[m * 64 + h * 16 + n] = (_Float16)v; });
        __syncthreads();
    }

    for (int i = tid; i < EE * EE; i += 256) WO[i] = (_Float16)woutw[i];
    __syncthreads();

    // out = o @ Wout^T + b ; accumulate token-mean per feature
    gemm_wmma(O, 64, WO, 64, 64, 64, 64, 64,
              [&](int m, int n, float v) { atomicAdd(&fmean[n], v + woutb[n]); });
    __syncthreads();
    if (tid < EE) feat[(size_t)b * EE + tid] = fmean[tid] * (1.0f / 64.0f);
}

// ---------------------------------------------------------------------------
extern "C" void kernel_launch(void* const* d_in, const int* in_sizes, int n_in,
                              void* d_out, int out_size, void* d_ws, size_t ws_size,
                              hipStream_t stream) {
    const float* eeg   = (const float*)d_in[0];
    const float* w1    = (const float*)d_in[1];
    const float* b1v   = (const float*)d_in[2];
    const float* w2    = (const float*)d_in[3];
    const float* b2v   = (const float*)d_in[4];
    const float* bn1g  = (const float*)d_in[5];
    const float* bn1b  = (const float*)d_in[6];
    const float* bn2g  = (const float*)d_in[7];
    const float* bn2b  = (const float*)d_in[8];
    const float* winw  = (const float*)d_in[9];
    const float* winb  = (const float*)d_in[10];
    const float* woutw = (const float*)d_in[11];
    const float* woutb = (const float*)d_in[12];

    float* out      = (float*)d_out;         // features [128*64] then wpli [128*64*64]
    float* out_wpli = out + BATCH * EE;

    char*     ws     = (char*)d_ws;
    float*    stats1 = (float*)ws;                         // 128 f32
    float*    stats2 = (float*)(ws + 512);                 // 128 f32
    _Float16* adj    = (_Float16*)(ws + 1024);             // 128 * 64*64 f16 (1 MB)
    _Float16* y1     = (_Float16*)(ws + 1024 + (size_t)BATCH * CCH * CCH * 2);
    _Float16* y2     = (_Float16*)((char*)y1 + (size_t)BATCH * CCH * HID * 2);

    k0_zero<<<1, 256, 0, stream>>>(stats1);
    k1_wpli_gcn1<<<BATCH, 256, 0, stream>>>(eeg, w1, b1v, out_wpli, adj, y1, stats1);
    k2_gcn2<<<BATCH, 256, 0, stream>>>(w2, b2v, bn1g, bn1b, adj, y1, y2, stats1, stats2);
    k3_mha<<<BATCH, 256, 0, stream>>>(winw, winb, woutw, woutb, bn2g, bn2b, y2, stats2, out);
}
